// Attention_28681791603418
// MI455X (gfx1250) — matrix operand order repaired and checked by reading, not compiled, not run on hardware
//
#include <hip/hip_runtime.h>
#include <hip/hip_bf16.h>
#include <hip/hip_fp16.h>

// MI455X / gfx1250, wave32. f16 WMMA (16x16x32) with f32 accumulation.

typedef _Float16 h16;
typedef __attribute__((ext_vector_type(16))) _Float16 v16h;
typedef __attribute__((ext_vector_type(8)))  _Float16 v8h;
typedef __attribute__((ext_vector_type(4)))  _Float16 v4h;
typedef __attribute__((ext_vector_type(8)))  float    v8f;
typedef __attribute__((ext_vector_type(4)))  float    v4f;

#define NB     4
#define SEQ    2048
#define DMODEL 1024
#define DHEAD  64

// ---- helpers ---------------------------------------------------------------

// A-operand (16x32 f16): lane m = lane&15, hh = lane>>4 holds
// K = {8*hh .. 8*hh+7} in elements 0..7 and {16+8*hh ..} in elements 8..15.
__device__ __forceinline__ v16h load_a16(const h16* __restrict__ p, int hh) {
  union { v16h v; v8h p8[2]; } u;
  u.p8[0] = *(const v8h*)(p + 8 * hh);
  u.p8[1] = *(const v8h*)(p + 16 + 8 * hh);
  return u.v;
}

__device__ __forceinline__ v8f wmma16(v16h a, v16h b, v8f c) {
  return __builtin_amdgcn_wmma_f32_16x16x32_f16(false, a, false, b, (short)0, c,
                                                false, false);
}

// ds_swizzle XOR-butterfly step (group-of-32 mode: offset = xor<<10 | and=0x1f).
// Single ds_swizzle_b32, no index math / bounds checks (vs. __shfl_xor ->
// ds_bpermute + ~5 VALU). XOR masks <= 8 stay within each 16-lane half, which
// matches the WMMA C-layout (a row's 16 columns live in one half).
template <int XM>
__device__ __forceinline__ float swz_xor(float v) {
  int s = __builtin_amdgcn_ds_swizzle(__float_as_int(v), (XM << 10) | 0x1f);
  return __int_as_float(s);
}
__device__ __forceinline__ float rowmax16(float v) {
  v = fmaxf(v, swz_xor<1>(v));
  v = fmaxf(v, swz_xor<2>(v));
  v = fmaxf(v, swz_xor<4>(v));
  v = fmaxf(v, swz_xor<8>(v));
  return v;
}
__device__ __forceinline__ float rowsum16(float v) {
  v += swz_xor<1>(v);
  v += swz_xor<2>(v);
  v += swz_xor<4>(v);
  v += swz_xor<8>(v);
  return v;
}

// ---- fp32 -> fp16 conversion (vectorized x4) -------------------------------

__global__ __launch_bounds__(256) void cvt_f32_f16(const float* __restrict__ in,
                                                   h16* __restrict__ out,
                                                   int n4) {
  int i = blockIdx.x * blockDim.x + threadIdx.x;
  if (i < n4) {
    v4f v = *(const v4f*)(in + 4 * (size_t)i);
    v4h o;
    o.x = (h16)v.x; o.y = (h16)v.y; o.z = (h16)v.z; o.w = (h16)v.w;
    *(v4h*)(out + 4 * (size_t)i) = o;
  }
}

// ---- GEMM: Y = f16( X . W^T + bias ), 32x64 tile per wave ------------------
// Per k-step: 2 A loads + 4 B loads feed 8 WMMAs (A reused 4x, B reused 2x).
// TRANSPOSED=true stores Yt[b][n][s] (packed 16B stores along s).

template <bool TRANSPOSED>
__global__ __launch_bounds__(32) void gemm_xwT(const h16* __restrict__ X,
                                               const h16* __restrict__ W,
                                               const float* __restrict__ bias,
                                               h16* __restrict__ Y,
                                               int N, int K, int Srows) {
  const int m0 = blockIdx.x * 32;
  const int n0 = blockIdx.y * 64;
  const int lane = threadIdx.x;
  const int mr = lane & 15, hh = lane >> 4;

  const h16* xrow0 = X + (size_t)(m0 + mr) * K;
  const h16* xrow1 = X + (size_t)(m0 + 16 + mr) * K;
  const h16* wrow0 = W + (size_t)(n0 + mr) * K;
  const h16* wrow1 = W + (size_t)(n0 + 16 + mr) * K;
  const h16* wrow2 = W + (size_t)(n0 + 32 + mr) * K;
  const h16* wrow3 = W + (size_t)(n0 + 48 + mr) * K;

  v8f acc[2][4];
  for (int i = 0; i < 2; ++i)
    for (int t = 0; t < 4; ++t) acc[i][t] = (v8f){};

  for (int k0 = 0; k0 < K; k0 += 32) {
    v16h a0 = load_a16(xrow0 + k0, hh);
    v16h a1 = load_a16(xrow1 + k0, hh);
    v16h b0 = load_a16(wrow0 + k0, hh);
    v16h b1 = load_a16(wrow1 + k0, hh);
    v16h b2 = load_a16(wrow2 + k0, hh);
    v16h b3 = load_a16(wrow3 + k0, hh);
    acc[0][0] = wmma16(a0, b0, acc[0][0]);
    acc[1][0] = wmma16(a1, b0, acc[1][0]);
    acc[0][1] = wmma16(a0, b1, acc[0][1]);
    acc[1][1] = wmma16(a1, b1, acc[1][1]);
    acc[0][2] = wmma16(a0, b2, acc[0][2]);
    acc[1][2] = wmma16(a1, b2, acc[1][2]);
    acc[0][3] = wmma16(a0, b3, acc[0][3]);
    acc[1][3] = wmma16(a1, b3, acc[1][3]);
  }

  if (TRANSPOSED) {
    // 32-row tile never crosses a batch (SEQ % 32 == 0)
    const int b = m0 / Srows;
    const int sbase = m0 % Srows;
    for (int t = 0; t < 4; ++t) {
      const int col = n0 + 16 * t + mr;
      const float bv = bias[col];
      h16* dst = Y + ((size_t)b * N + col) * Srows + sbase;
      for (int i = 0; i < 2; ++i) {
        v8h pack;
        for (int r = 0; r < 8; ++r) pack[r] = (h16)(acc[i][t][r] + bv);
        *(v8h*)(dst + 16 * i + 8 * hh) = pack;
      }
    }
  } else {
    for (int t = 0; t < 4; ++t) {
      const int col = n0 + 16 * t + mr;
      const float bv = bias[col];
      for (int i = 0; i < 2; ++i)
        for (int r = 0; r < 8; ++r)
          Y[(size_t)(m0 + 16 * i + r + 8 * hh) * N + col] =
              (h16)(acc[i][t][r] + bv);
    }
  }
}

// ---- flash attention -------------------------------------------------------
// Grid: (SEQ/16, NB). Block: 256 threads (8 waves).
// Software-pipelined: wave 0 produces P/scale for key-tile i+1 into LDS buffer
// (i+1)&1 while all waves consume buffer i&1 (rescale + 8 PxV WMMAs over this
// wave's 128 output columns). One barrier per iteration; double buffering
// makes the overwrite safe (buffer written at iter i+1 was freed by the
// barrier ending iter i).

__global__ __launch_bounds__(256) void attn_flash(const h16* __restrict__ Qh,
                                                  const h16* __restrict__ Kh,
                                                  const h16* __restrict__ Vt,
                                                  float* __restrict__ out) {
  __shared__ __align__(64) h16 pLds[2][16][32];   // P tiles, row-major
  __shared__ float scaleLds[2][16];
  __shared__ float lLds[16];

  const int q0 = blockIdx.x * 16;
  const int batch = blockIdx.y;
  const int tid = threadIdx.x;
  const int w = tid >> 5;
  const int lane = tid & 31;
  const int mr = lane & 15, hh = lane >> 4;

  const h16* Qb = Qh + (size_t)batch * SEQ * DHEAD;
  const h16* Kb = Kh + (size_t)batch * SEQ * DHEAD;
  const h16* Vb = Vt + (size_t)batch * DMODEL * SEQ;   // [e][s]

  v8f o[8];
  for (int j = 0; j < 8; ++j) o[j] = (v8f){};

  v16h a0 = {}, a1 = {};
  float rmax[8], lsum[8];
  for (int r = 0; r < 8; ++r) { rmax[r] = -1e30f; lsum[r] = 0.0f; }

  if (w == 0) {
    const h16* qrow = Qb + (size_t)(q0 + mr) * DHEAD;
    a0 = load_a16(qrow, hh);        // d = 0..31
    a1 = load_a16(qrow + 32, hh);   // d = 32..63
  }

  // produce scores+softmax for key tile starting at k0 into LDS buffer `buf`
  auto produce = [&](int k0, int buf) {
    v8f c[2];
    for (int t = 0; t < 2; ++t) {
      const h16* krow = Kb + (size_t)(k0 + 16 * t + mr) * DHEAD;
      v16h b0 = load_a16(krow, hh);        // d 0..31
      v16h b1 = load_a16(krow + 32, hh);   // d 32..63
      v8f cc = {};
      cc = wmma16(a0, b0, cc);
      cc = wmma16(a1, b1, cc);
      c[t] = cc;
    }
    for (int r = 0; r < 8; ++r) {
      const int qrow = q0 + r + 8 * hh;
      if (k0 + mr > qrow)      c[0][r] = -1e30f;       // causal mask
      if (k0 + 16 + mr > qrow) c[1][r] = -1e30f;
      const float mt = rowmax16(fmaxf(c[0][r], c[1][r]));
      const float mnew = fmaxf(rmax[r], mt);
      const float scale = __expf(rmax[r] - mnew);
      const float p0 = __expf(c[0][r] - mnew);
      const float p1 = __expf(c[1][r] - mnew);
      lsum[r] = lsum[r] * scale + rowsum16(p0 + p1);
      rmax[r] = mnew;
      pLds[buf][r + 8 * hh][mr] = (h16)p0;
      pLds[buf][r + 8 * hh][16 + mr] = (h16)p1;
      if (mr == 0) scaleLds[buf][r + 8 * hh] = scale;
    }
  };

  const int kend = q0 + 16;                 // keys <= q (exclusive bound)
  const int nIter = (kend + 31) / 32;

  if (w == 0) produce(0, 0);
  __syncthreads();

  for (int it = 0; it < nIter; ++it) {
    const int k0 = it * 32;
    const int buf = it & 1;
    if (w == 0 && it + 1 < nIter) produce(k0 + 32, buf ^ 1);

    float sc[8];
    for (int r = 0; r < 8; ++r) sc[r] = scaleLds[buf][r + 8 * hh];
    union { v16h v; v8h p8[2]; } pa;        // P as A-operand (16x32)
    pa.p8[0] = *(const v8h*)&pLds[buf][mr][8 * hh];
    pa.p8[1] = *(const v8h*)&pLds[buf][mr][16 + 8 * hh];
    const bool more = (it + 1) < nIter;
    for (int j = 0; j < 8; ++j) {
      const int col = w * 128 + j * 16 + mr;
      const h16* vrow = Vb + (size_t)col * SEQ + k0;
      v16h vb = load_a16(vrow, hh);   // two 16B chunks, table k order (V transposed)
      v8f oo = o[j];
      for (int r = 0; r < 8; ++r) oo[r] *= sc[r];
      o[j] = wmma16(pa.v, vb, oo);
      if (more) __builtin_prefetch(vrow + 32, 0, 1);  // global_prefetch_b8
    }
    __syncthreads();
  }

  if (w == 0 && mr == 0)
    for (int r = 0; r < 8; ++r) lLds[r + 8 * hh] = lsum[r];
  __syncthreads();

  float inv[8];
  for (int r = 0; r < 8; ++r) inv[r] = 1.0f / lLds[r + 8 * hh];
  float* ob = out + ((size_t)batch * SEQ + q0) * DMODEL;
  for (int j = 0; j < 8; ++j) {
    const int col = w * 128 + j * 16 + mr;
    for (int r = 0; r < 8; ++r)
      ob[(size_t)(r + 8 * hh) * DMODEL + col] = o[j][r] * inv[r];
  }
}

// ---- host ------------------------------------------------------------------

extern "C" void kernel_launch(void* const* d_in, const int* in_sizes, int n_in,
                              void* d_out, int out_size, void* d_ws,
                              size_t ws_size, hipStream_t stream) {
  const float* x   = (const float*)d_in[0];   // [4,2048,1024]
  const float* WQw = (const float*)d_in[1];   // [64,1024]
  const float* WQb = (const float*)d_in[2];   // [64]
  const float* WKw = (const float*)d_in[3];   // [64,1024]
  const float* WKb = (const float*)d_in[4];   // [64]
  const float* WVw = (const float*)d_in[5];   // [1024,1024]
  const float* WVb = (const float*)d_in[6];   // [1024]
  float* out = (float*)d_out;

  const size_t nX  = (size_t)NB * SEQ * DMODEL;      // 8388608
  const size_t nWq = (size_t)DHEAD * DMODEL;         // 65536
  const size_t nWv = (size_t)DMODEL * DMODEL;        // 1048576
  const size_t nQK = (size_t)NB * SEQ * DHEAD;       // 524288

  char* ws = (char*)d_ws;
  size_t off = 0;
  auto alloc_h = [&](size_t elems) {
    h16* p = (h16*)(ws + off);
    off += ((elems * sizeof(h16)) + 255) & ~(size_t)255;
    return p;
  };
  h16* xh  = alloc_h(nX);
  h16* wqh = alloc_h(nWq);
  h16* wkh = alloc_h(nWq);
  h16* wvh = alloc_h(nWv);
  h16* qh  = alloc_h(nQK);
  h16* kh  = alloc_h(nQK);
  h16* vt  = alloc_h(nX);          // transposed V: [b][e][s]

  // fp32 -> fp16
  cvt_f32_f16<<<(int)(nX / 4 + 255) / 256, 256, 0, stream>>>(x, xh, (int)(nX / 4));
  cvt_f32_f16<<<(int)(nWq / 4 + 255) / 256, 256, 0, stream>>>(WQw, wqh, (int)(nWq / 4));
  cvt_f32_f16<<<(int)(nWq / 4 + 255) / 256, 256, 0, stream>>>(WKw, wkh, (int)(nWq / 4));
  cvt_f32_f16<<<(int)(nWv / 4 + 255) / 256, 256, 0, stream>>>(WVw, wvh, (int)(nWv / 4));

  const int Mtiles = (NB * SEQ) / 32;   // 256
  // Q, K projections (row-major f16 outputs); N=64 -> one 64-wide column tile
  gemm_xwT<false><<<dim3(Mtiles, DHEAD / 64), 32, 0, stream>>>(
      xh, wqh, WQb, qh, DHEAD, DMODEL, SEQ);
  gemm_xwT<false><<<dim3(Mtiles, DHEAD / 64), 32, 0, stream>>>(
      xh, wkh, WKb, kh, DHEAD, DMODEL, SEQ);
  // V projection, stored transposed per batch: vt[b][e][s]
  gemm_xwT<true><<<dim3(Mtiles, DMODEL / 64), 32, 0, stream>>>(
      xh, wvh, WVb, vt, DMODEL, DMODEL, SEQ);
  // causal flash attention
  attn_flash<<<dim3(SEQ / 16, NB), 256, 0, stream>>>(qh, kh, vt, out);
}
